// Attention_18854906430172
// MI455X (gfx1250) — matrix operand order repaired and checked by reading, not compiled, not run on hardware
//
#include <hip/hip_runtime.h>

typedef __attribute__((ext_vector_type(16))) _Float16 v16h;
typedef __attribute__((ext_vector_type(8)))  float    v8f;

#define C_DIM   64
#define N_TOK   4096   // H*W per batch
#define NH      4
#define HEADD   16
#define NEG_SLOPE 0.01f
#define LN_EPS  1e-5f
#define SM_SCALE 0.25f   // 1/sqrt(16), folded into qaT

static __device__ inline v8f wmma16x16x32(v16h a, v16h b, v8f c) {
    return __builtin_amdgcn_wmma_f32_16x16x32_f16(
        false, a, false, b, (short)0, c, false, false);
}

// A-frag (16x32, f16) from row-major MxK matrix, row stride ld, K offset kofs.
// ISA 7.12.2: lane L: M = L&15; lanes<16 hold K = kofs+{0..7,16..23},
// lanes>=16 hold K = kofs+{8..15,24..31}.
static __device__ inline v16h a_frag_rm(const _Float16* base, int ld, int kofs) {
    int l = threadIdx.x & 31;
    int m = l & 15;
    int kb = (l >> 4) * 8;
    const _Float16* r = base + m * ld + kofs + kb;
    v16h a;
#pragma unroll
    for (int j = 0; j < 8; ++j) a[j] = r[j];
#pragma unroll
    for (int j = 0; j < 8; ++j) a[8 + j] = r[16 + j];
    return a;
}

// A-frag with K-dim = 16 zero-padded to 32: kp points at (row, kb) of a
// row-major Mx16 tile; elements 8..15 (K = 16..31) are zero.
static __device__ inline v16h a_frag_k16(const _Float16* kp) {
    v16h a;
#pragma unroll
    for (int j = 0; j < 8; ++j) a[j] = kp[j];
#pragma unroll
    for (int j = 0; j < 8; ++j) a[8 + j] = (_Float16)0.f;
    return a;
}

// B-frag (32x16, f16) from row-major KxN matrix: lane L holds column
// ncol0+(L&15), K = krow0+{0..7,16..23} (L<16) or krow0+{8..15,24..31}.
static __device__ inline v16h b_frag_rm(const _Float16* base, int ld, int krow0, int ncol0) {
    int l = threadIdx.x & 31;
    const _Float16* r = base + (krow0 + (l >> 4) * 8) * ld + ncol0 + (l & 15);
    v16h b;
#pragma unroll
    for (int j = 0; j < 8; ++j) b[j] = r[j * ld];
#pragma unroll
    for (int j = 0; j < 8; ++j) b[8 + j] = r[(16 + j) * ld];
    return b;
}

// ---------------------------------------------------------------------------
// Kernel 1: LayerNorm + MLP(64->128 LeakyReLU ->64) for q/k/v projections.
// in: [B, C, HW] f32.
//   out_h : f16 [B*nh, N, 16]   token-major  (k, v)      scaled by hscale
//   out_hT: f16 [B*nh, 16, N]   d-major      (q)         scaled by hscale
//   out_tok: f32 [B*N, C]       (v only, residual)
// ---------------------------------------------------------------------------
__global__ __launch_bounds__(128) void qkv_proj_kernel(
    const float* __restrict__ in,
    const float* __restrict__ ln_g, const float* __restrict__ ln_b,
    const float* __restrict__ w1, const float* __restrict__ b1,
    const float* __restrict__ w2, const float* __restrict__ b2,
    _Float16* __restrict__ out_h, _Float16* __restrict__ out_hT,
    float hscale, float* __restrict__ out_tok)
{
    __shared__ __align__(16) _Float16 Xh[64 * 64];
    __shared__ __align__(16) _Float16 W1T[64 * 128];
    __shared__ __align__(16) _Float16 W2T[128 * 64];
    __shared__ __align__(16) _Float16 Hh[4][16 * 128];
    __shared__ float b1s[128];
    __shared__ float b2s[64];

    const int tid = threadIdx.x;
    const int chunk = blockIdx.x;   // 64 tokens each

    for (int j = tid; j < 64 * 128; j += 128) {
        int i = j >> 7, o = j & 127;
        W1T[j] = (_Float16)w1[o * 64 + i];
    }
    for (int j = tid; j < 128 * 64; j += 128) {
        int i = j >> 6, o = j & 63;
        W2T[j] = (_Float16)w2[o * 128 + i];
    }
    if (tid < 128) b1s[tid] = b1[tid];
    if (tid < 64)  b2s[tid] = b2[tid];

    // LayerNorm: pair of adjacent lanes handles one token (32 channels each)
    {
        int token = tid >> 1;
        int half  = tid & 1;
        int g  = chunk * 64 + token;
        int bb = g >> 12;
        int n  = g & (N_TOK - 1);
        const float* src = in + (size_t)bb * C_DIM * N_TOK + n;
        float xv[32];
        float s = 0.f, ss = 0.f;
#pragma unroll
        for (int j = 0; j < 32; ++j) {
            float v = src[(size_t)(half * 32 + j) * N_TOK];
            xv[j] = v; s += v; ss += v * v;
        }
        s  += __shfl_xor(s, 1);
        ss += __shfl_xor(ss, 1);
        float mean = s * (1.f / 64.f);
        float var  = ss * (1.f / 64.f) - mean * mean;
        float rstd = rsqrtf(var + LN_EPS);
#pragma unroll
        for (int j = 0; j < 32; ++j) {
            int c = half * 32 + j;
            float yv = (xv[j] - mean) * rstd * ln_g[c] + ln_b[c];
            Xh[token * 64 + c] = (_Float16)yv;
        }
    }
    __syncthreads();

    const int wave = tid >> 5;
    const int lane = tid & 31;
    const int colsel = lane & 15;
    const int rowadd = (lane >> 4) * 8;
    _Float16* Xw = &Xh[wave * 16 * 64];
    _Float16* Hw = Hh[wave];

    // GEMM1: [16x64] x [64x128]
    v16h a0 = a_frag_rm(Xw, 64, 0);
    v16h a1 = a_frag_rm(Xw, 64, 32);
#pragma unroll
    for (int nt = 0; nt < 8; ++nt) {
        v8f c = {0.f,0.f,0.f,0.f,0.f,0.f,0.f,0.f};
        v16h bk0 = b_frag_rm(W1T, 128, 0,  nt * 16);
        c = wmma16x16x32(a0, bk0, c);
        v16h bk1 = b_frag_rm(W1T, 128, 32, nt * 16);
        c = wmma16x16x32(a1, bk1, c);
#pragma unroll
        for (int e = 0; e < 8; ++e) {
            float h = c[e] + b1s[nt * 16 + colsel];
            h = h > 0.f ? h : NEG_SLOPE * h;
            Hw[(e + rowadd) * 128 + nt * 16 + colsel] = (_Float16)h;
        }
    }
    asm volatile("s_wait_dscnt 0" ::: "memory");  // wave-local LDS RAW

    // GEMM2: [16x128] x [128x64]
    const int g0 = chunk * 64 + wave * 16;
#pragma unroll
    for (int nt = 0; nt < 4; ++nt) {
        v8f c = {0.f,0.f,0.f,0.f,0.f,0.f,0.f,0.f};
#pragma unroll
        for (int kt = 0; kt < 4; ++kt) {
            v16h a = a_frag_rm(Hw, 128, kt * 32);
            v16h b = b_frag_rm(W2T, 64, kt * 32, nt * 16);
            c = wmma16x16x32(a, b, c);
        }
#pragma unroll
        for (int e = 0; e < 8; ++e) {
            int gg  = g0 + e + rowadd;
            int bb  = gg >> 12;
            int tok = gg & (N_TOK - 1);
            int cidx = nt * 16 + colsel;         // head = nt, d = colsel
            float y = c[e] + b2s[cidx];
            if (out_h)
                out_h[(((size_t)bb * NH + nt) * N_TOK + tok) * HEADD + colsel] =
                    (_Float16)(y * hscale);
            if (out_hT)   // d-major [bh][d][N]
                out_hT[(((size_t)bb * NH + nt) * HEADD + colsel) * N_TOK + tok] =
                    (_Float16)(y * hscale);
            if (out_tok) out_tok[(size_t)gg * C_DIM + cidx] = y;
        }
    }
}

// ---------------------------------------------------------------------------
// Kernel 2: flash attention, transposed-S formulation. Grid (N/128, B*nh),
// 8 waves/WG, 16 queries/wave, no LDS, no barriers.
//   qaT: f16 [B*nh, 16, N] (pre-scaled by 1/sqrt(d))
//   ka, va: f16 [B*nh, N, 16]
//   xout: f32 [B*N, C] token-major
// S^T = K*Q^T puts one query per lane (col), and its C-layout element map
// (key = e + 8*(lane>>4)) equals the A-frag layout P needs for O = P*V,
// so P converts C->A entirely in registers.
// ---------------------------------------------------------------------------
__global__ __launch_bounds__(256) void flash_attn_kernel(
    const _Float16* __restrict__ qaT, const _Float16* __restrict__ ka,
    const _Float16* __restrict__ va, float* __restrict__ xout)
{
    const int tid  = threadIdx.x;
    const int lane = tid & 31;
    const int wave = tid >> 5;
    const int bh = blockIdx.y;              // b*4 + h
    const int bb = bh >> 2;
    const int hh = bh & 3;
    const int qt0 = blockIdx.x * 128 + wave * 16;
    const int r  = lane & 15;
    const int kb = (lane >> 4) * 8;

    // B = Q^T tile: lane = query column (lane&15), elements 0..7 = d kb..kb+7,
    // elements 8..15 (d 16..31) zero-padded. Loaded once, reused for all keys.
    v16h bq;
    {
        const _Float16* qp = qaT + ((size_t)bh * HEADD + kb) * N_TOK + qt0 + r;
#pragma unroll
        for (int e = 0; e < 8; ++e) bq[e] = qp[(size_t)e * N_TOK];
#pragma unroll
        for (int e = 0; e < 8; ++e) bq[8 + e] = (_Float16)0.f;
    }

    float m_run = -1e30f, l_run = 0.f;
    v8f co = {0.f,0.f,0.f,0.f,0.f,0.f,0.f,0.f};
    const size_t kvb = (size_t)bh * N_TOK * HEADD;

    for (int blk = 0; blk < N_TOK / 32; ++blk) {
        // K tile rows direct from global (K/V stay L2-resident: 2 MB total)
        const _Float16* kp0 = ka + kvb + (size_t)(blk * 32 + r) * HEADD + kb;
        v16h ak0 = a_frag_k16(kp0);
        v16h ak1 = a_frag_k16(kp0 + 16 * HEADD);
        v8f cT0 = {0.f,0.f,0.f,0.f,0.f,0.f,0.f,0.f};
        v8f cT1 = {0.f,0.f,0.f,0.f,0.f,0.f,0.f,0.f};
        cT0 = wmma16x16x32(ak0, bq, cT0);   // S^T keys 0..15
        cT1 = wmma16x16x32(ak1, bq, cT1);   // S^T keys 16..31

        if (blk + 1 < N_TOK / 32) {
            if (lane < 8)
                __builtin_prefetch(ka + kvb + (size_t)(blk + 1) * 512 + lane * 64, 0, 0);
            else if (lane < 16)
                __builtin_prefetch(va + kvb + (size_t)(blk + 1) * 512 + (lane - 8) * 64, 0, 0);
        }

        // lane holds 16 scores of query (lane&15); partner lane^16 holds the
        // complementary 16 keys -> per-lane tree + one shfl_xor(16).
        float bm = cT0[0];
#pragma unroll
        for (int e = 0; e < 8; ++e) { bm = fmaxf(bm, cT0[e]); bm = fmaxf(bm, cT1[e]); }
        bm = fmaxf(bm, __shfl_xor(bm, 16));
        float mnew = fmaxf(m_run, bm);
        float corr = __expf(m_run - mnew);

        float p0[8], p1[8];
        float rs = 0.f;
#pragma unroll
        for (int e = 0; e < 8; ++e) {
            p0[e] = __expf(cT0[e] - mnew);
            p1[e] = __expf(cT1[e] - mnew);
            rs += p0[e] + p1[e];
        }
        rs += __shfl_xor(rs, 16);
        l_run = l_run * corr + rs;
        m_run = mnew;

        // P: C-layout of S^T == A-frag layout (m = lane&15 = query,
        // K = kb+e / 16+kb+e = key) -> register-only conversion.
        v16h ap;
#pragma unroll
        for (int e = 0; e < 8; ++e) {
            ap[e]     = (_Float16)p0[e];
            ap[8 + e] = (_Float16)p1[e];
        }

        // rescale O accumulator: corr per query, O rows are queries e+kb
#pragma unroll
        for (int e = 0; e < 8; ++e) co[e] *= __shfl(corr, e + kb);

        const _Float16* vp = va + kvb + (size_t)(blk * 32 + kb) * HEADD + r;
        v16h bv;
#pragma unroll
        for (int e = 0; e < 8; ++e) bv[e] = vp[e * HEADD];
#pragma unroll
        for (int e = 0; e < 8; ++e) bv[8 + e] = vp[(16 + e) * HEADD];
        co = wmma16x16x32(ap, bv, co);      // O += P * V
    }

    float linv = 1.f / l_run;
#pragma unroll
    for (int e = 0; e < 8; ++e) {
        float o = co[e] * __shfl(linv, e + kb);
        int tok = qt0 + e + kb;
        xout[(((size_t)bb * N_TOK) + tok) * C_DIM + hh * HEADD + r] = o;
    }
}

// ---------------------------------------------------------------------------
// Kernel 3: residual MLP (m1/m2): out = res + MLP(in). Token-major f32 I/O.
// out_tok: f32 [B*N, C] (nullable). out_chw: f32 [B, C, HW] (nullable).
// ---------------------------------------------------------------------------
__global__ __launch_bounds__(128) void post_mlp_kernel(
    const float* __restrict__ in_tok, const float* __restrict__ res_tok,
    const float* __restrict__ w1, const float* __restrict__ b1,
    const float* __restrict__ w2, const float* __restrict__ b2,
    float* __restrict__ out_tok, float* __restrict__ out_chw)
{
    __shared__ __align__(16) _Float16 Xh[64 * 64];
    __shared__ __align__(16) _Float16 W1T[64 * 128];
    __shared__ __align__(16) _Float16 W2T[128 * 64];
    __shared__ __align__(16) _Float16 Hh[4][16 * 128];
    __shared__ float b1s[128];
    __shared__ float b2s[64];

    const int tid = threadIdx.x;
    const int chunk = blockIdx.x;

    for (int j = tid; j < 64 * 128; j += 128) {
        int i = j >> 7, o = j & 127;
        W1T[j] = (_Float16)w1[o * 64 + i];
    }
    for (int j = tid; j < 128 * 64; j += 128) {
        int i = j >> 6, o = j & 63;
        W2T[j] = (_Float16)w2[o * 128 + i];
    }
    if (tid < 128) b1s[tid] = b1[tid];
    if (tid < 64)  b2s[tid] = b2[tid];

    for (int j = tid; j < 64 * 64; j += 128)
        Xh[j] = (_Float16)in_tok[(size_t)chunk * 4096 + j];
    __syncthreads();

    const int wave = tid >> 5;
    const int lane = tid & 31;
    const int colsel = lane & 15;
    const int rowadd = (lane >> 4) * 8;
    _Float16* Xw = &Xh[wave * 16 * 64];
    _Float16* Hw = Hh[wave];

    v16h a0 = a_frag_rm(Xw, 64, 0);
    v16h a1 = a_frag_rm(Xw, 64, 32);
#pragma unroll
    for (int nt = 0; nt < 8; ++nt) {
        v8f c = {0.f,0.f,0.f,0.f,0.f,0.f,0.f,0.f};
        v16h bk0 = b_frag_rm(W1T, 128, 0,  nt * 16);
        c = wmma16x16x32(a0, bk0, c);
        v16h bk1 = b_frag_rm(W1T, 128, 32, nt * 16);
        c = wmma16x16x32(a1, bk1, c);
#pragma unroll
        for (int e = 0; e < 8; ++e) {
            float h = c[e] + b1s[nt * 16 + colsel];
            h = h > 0.f ? h : NEG_SLOPE * h;
            Hw[(e + rowadd) * 128 + nt * 16 + colsel] = (_Float16)h;
        }
    }
    asm volatile("s_wait_dscnt 0" ::: "memory");

    const int g0 = chunk * 64 + wave * 16;
#pragma unroll
    for (int nt = 0; nt < 4; ++nt) {
        v8f c = {0.f,0.f,0.f,0.f,0.f,0.f,0.f,0.f};
#pragma unroll
        for (int kt = 0; kt < 4; ++kt) {
            v16h a = a_frag_rm(Hw, 128, kt * 32);
            v16h b = b_frag_rm(W2T, 64, kt * 32, nt * 16);
            c = wmma16x16x32(a, b, c);
        }
#pragma unroll
        for (int e = 0; e < 8; ++e) {
            int gg  = g0 + e + rowadd;
            int bb  = gg >> 12;
            int tok = gg & (N_TOK - 1);
            int cidx = nt * 16 + colsel;
            float y = c[e] + b2s[cidx] + res_tok[(size_t)gg * C_DIM + cidx];
            if (out_tok) out_tok[(size_t)gg * C_DIM + cidx] = y;
            if (out_chw) out_chw[((size_t)bb * C_DIM + cidx) * N_TOK + tok] = y;
        }
    }
}

// ---------------------------------------------------------------------------
extern "C" void kernel_launch(void* const* d_in, const int* in_sizes, int n_in,
                              void* d_out, int out_size, void* d_ws, size_t ws_size,
                              hipStream_t stream) {
    const float* q = (const float*)d_in[0];
    const float* k = (const float*)d_in[1];
    const float* v = (const float*)d_in[2];
    const float* q_ln_g = (const float*)d_in[3],  *q_ln_b = (const float*)d_in[4];
    const float* q_w1 = (const float*)d_in[5],    *q_b1 = (const float*)d_in[6];
    const float* q_w2 = (const float*)d_in[7],    *q_b2 = (const float*)d_in[8];
    const float* k_ln_g = (const float*)d_in[9],  *k_ln_b = (const float*)d_in[10];
    const float* k_w1 = (const float*)d_in[11],   *k_b1 = (const float*)d_in[12];
    const float* k_w2 = (const float*)d_in[13],   *k_b2 = (const float*)d_in[14];
    const float* v_ln_g = (const float*)d_in[15], *v_ln_b = (const float*)d_in[16];
    const float* v_w1 = (const float*)d_in[17],   *v_b1 = (const float*)d_in[18];
    const float* v_w2 = (const float*)d_in[19],   *v_b2 = (const float*)d_in[20];
    const float* m1_w1 = (const float*)d_in[21],  *m1_b1 = (const float*)d_in[22];
    const float* m1_w2 = (const float*)d_in[23],  *m1_b2 = (const float*)d_in[24];
    const float* m2_w1 = (const float*)d_in[25],  *m2_b1 = (const float*)d_in[26];
    const float* m2_w2 = (const float*)d_in[27],  *m2_b2 = (const float*)d_in[28];

    char* ws = (char*)d_ws;
    // f16 buffers, 1 MB each: qaT [B*nh,16,N]; ka,va [B*nh,N,16]
    _Float16* qaT = (_Float16*)(ws + 0);
    _Float16* ka  = (_Float16*)(ws + (1u << 20));
    _Float16* va  = (_Float16*)(ws + (2u << 20));
    // token-major f32 buffers [B*N, C]: 2 MB each
    float* va1 = (float*)(ws + (3u << 20));
    float* xat = (float*)(ws + (5u << 20));
    float* rs1 = (float*)(ws + (7u << 20));

    qkv_proj_kernel<<<dim3(128), 128, 0, stream>>>(
        q, q_ln_g, q_ln_b, q_w1, q_b1, q_w2, q_b2,
        nullptr, qaT, SM_SCALE, nullptr);
    qkv_proj_kernel<<<dim3(128), 128, 0, stream>>>(
        k, k_ln_g, k_ln_b, k_w1, k_b1, k_w2, k_b2,
        ka, nullptr, 1.0f, nullptr);
    qkv_proj_kernel<<<dim3(128), 128, 0, stream>>>(
        v, v_ln_g, v_ln_b, v_w1, v_b1, v_w2, v_b2,
        va, nullptr, 1.0f, va1);

    flash_attn_kernel<<<dim3(32, 8), 256, 0, stream>>>(qaT, ka, va, xat);

    post_mlp_kernel<<<dim3(128), 128, 0, stream>>>(
        xat, va1, m1_w1, m1_b1, m1_w2, m1_b2, rs1, nullptr);
    post_mlp_kernel<<<dim3(128), 128, 0, stream>>>(
        rs1, rs1, m2_w1, m2_b1, m2_w2, m2_b2, nullptr, (float*)d_out);
}
